// AcceleratedGemma3nTextAttention_30081950941772
// MI455X (gfx1250) — matrix operand order repaired and checked by reading, not compiled, not run on hardware
//
#include <hip/hip_runtime.h>
#include <hip/hip_bf16.h>
#include <stdint.h>

// ---------------------------------------------------------------------------
// Problem constants (match the reference)
// ---------------------------------------------------------------------------
#define S_LEN 4096
#define D_MODEL 2048
#define N_HEADS 16
#define N_KV 4
#define HEAD_DIM 128
#define NQ (N_HEADS * HEAD_DIM)   // 2048
#define NKV (N_KV * HEAD_DIM)     // 512
#define SOFTMAX_SCALE 0.08838834764831845f  // 128^-0.5
#define LN10000 9.210340371976184f

typedef __bf16 v16bf __attribute__((ext_vector_type(16)));
typedef float  v8f   __attribute__((ext_vector_type(8)));
typedef unsigned short u16;

// ---------------------------------------------------------------------------
// Helpers
// ---------------------------------------------------------------------------
__device__ inline u16 f2b(float x) {               // f32 -> bf16 (RNE)
  unsigned int u = __float_as_uint(x);
  u += 0x7FFFu + ((u >> 16) & 1u);
  return (u16)(u >> 16);
}

union Frag { v16bf v; uint4 u[2]; };

// A-fragment (16x32, 16-bit): lane row = lane&15; halves 0..7 = K kb0..kb0+7,
// halves 8..15 = K kb1..kb1+7  (kb0 = lane<16?0:16, kb1 = lane<16?8:24)
__device__ inline v16bf load_frag_a(const u16* base, int kb0, int kb1) {
  Frag f;
  f.u[0] = *(const uint4*)(base + kb0);
  f.u[1] = *(const uint4*)(base + kb1);
  return f.v;
}
// B-fragment (32x16, 16-bit): lane col = lane&15; halves j = K kb+j, 16
// contiguous values (kb = lane<16?0:16) -> caller pre-offsets by kb.
__device__ inline v16bf load_frag_b(const u16* base) {
  Frag f;
  f.u[0] = *(const uint4*)(base);
  f.u[1] = *(const uint4*)(base + 8);
  return f.v;
}

__device__ inline v8f wmma_bf16(v16bf a, v16bf b, v8f c) {
  return __builtin_amdgcn_wmma_f32_16x16x32_bf16(false, a, false, b,
                                                 (short)0, c, false, false);
}

// ---------------------------------------------------------------------------
// Elementwise conversion kernels
// ---------------------------------------------------------------------------
__global__ void k_f32_to_bf16(const float* __restrict__ in, u16* __restrict__ out, int n) {
  int i = blockIdx.x * blockDim.x + threadIdx.x;
  if (i < n) out[i] = f2b(in[i]);
}

// W [K,N] f32 (row-major) -> Wt [N,K] bf16
__global__ void k_transpose_bf16(const float* __restrict__ W, u16* __restrict__ Wt,
                                 int K, int N) {
  int i = blockIdx.x * blockDim.x + threadIdx.x;
  if (i >= K * N) return;
  int k = i / N, n = i % N;
  Wt[(size_t)n * K + k] = f2b(W[i]);
}

// ---------------------------------------------------------------------------
// Fused QKV projection GEMM with RoPE / V-transpose epilogue.
// Wave tile = 16 rows x 128 cols = one full head -> RoPE pairs (d, d+64) are
// accumulators j and j+4 in the SAME lane (C-layout: col d -> acc d/16, lane
// d%16), so rotation is per-lane register math.
//   mode 0: Q -> RoPE + softmax-scale, Out = Qr [h][s][128] bf16
//   mode 1: K -> RoPE,                 Out = Kr [kv][s][128] bf16
//   mode 2: V -> transposed store,     Out = Vt [kv*128+d][s] bf16
// ---------------------------------------------------------------------------
__global__ void k_gemm_qkv(const u16* __restrict__ A, const u16* __restrict__ Bt,
                           const int* __restrict__ pos_ids, u16* __restrict__ Out,
                           int M, int nheads, int Kd, int mode) {
  int wid  = blockIdx.x * (blockDim.x >> 5) + (threadIdx.x >> 5);
  int lane = threadIdx.x & 31;
  int tm = wid / nheads, h = wid % nheads;
  if (tm * 16 >= M) return;
  int m0 = tm << 4;
  int n0 = h << 7;                // head base column
  int half16 = lane & 15;
  int hi  = (lane >= 16);
  int kb0 = hi ? 16 : 0, kb1 = hi ? 24 : 8;
  int kbB = hi ? 16 : 0;

  const u16* arow = A  + (size_t)(m0 + half16) * Kd;
  const u16* brow = Bt + (size_t)(n0 + half16) * Kd + kbB;

  v8f acc[8];
#pragma unroll
  for (int j = 0; j < 8; ++j) acc[j] = (v8f){};

  for (int kk = 0; kk < Kd; kk += 32) {
    __builtin_prefetch(arow + kk + 256, 0, 3);
    v16bf a = load_frag_a(arow + kk, kb0, kb1);
#pragma unroll
    for (int j = 0; j < 8; ++j)
      acc[j] = wmma_bf16(a, load_frag_b(brow + (size_t)(j * 16) * Kd + kk), acc[j]);
  }

  int mrow = m0 + (hi ? 8 : 0);
  if (mode == 2) {
    // V: store bf16 transposed -> Vt[(n0 + j*16 + half16) * S + s]
#pragma unroll
    for (int j = 0; j < 8; ++j) {
      u16* ocol = Out + (size_t)(n0 + j * 16 + half16) * S_LEN + mrow;
#pragma unroll
      for (int r = 0; r < 8; ++r) ocol[r] = f2b(acc[j][r]);
    }
  } else {
    float scale = (mode == 0) ? SOFTMAX_SCALE : 1.0f;
    float freq[4];
#pragma unroll
    for (int j = 0; j < 4; ++j) {
      int d = j * 16 + half16;    // pair index 0..63
      freq[j] = __expf(-((float)d * (1.0f / 64.0f)) * LN10000);  // 10000^{-d/64}
    }
#pragma unroll
    for (int r = 0; r < 8; ++r) {
      float p = (float)pos_ids[mrow + r];
      u16* orow = Out + ((size_t)h * S_LEN + mrow + r) * HEAD_DIM + half16;
#pragma unroll
      for (int j = 0; j < 4; ++j) {
        float sn, cs;
        __sincosf(p * freq[j], &sn, &cs);
        float x1 = acc[j][r], x2 = acc[j + 4][r];
        orow[j * 16]      = f2b((x1 * cs - x2 * sn) * scale);
        orow[j * 16 + 64] = f2b((x2 * cs + x1 * sn) * scale);
      }
    }
  }
}

// ---------------------------------------------------------------------------
// Generic WMMA GEMM: C[M,N] (f32, row-major) = A[M,K] * Bt[N,K]^T  (bf16)
// One wave computes a 16x64 tile (4 accumulators share one A fragment).
// Requires M%16==0, N%64==0, K%32==0.
// ---------------------------------------------------------------------------
__global__ void k_gemm_bf16(const u16* __restrict__ A, const u16* __restrict__ Bt,
                            float* __restrict__ C, int M, int N, int Kd) {
  int wid  = blockIdx.x * (blockDim.x >> 5) + (threadIdx.x >> 5);
  int lane = threadIdx.x & 31;
  int tilesN = N >> 6;
  int tm = wid / tilesN, tn = wid % tilesN;
  if (tm * 16 >= M) return;
  int m0 = tm << 4, n0 = tn << 6;
  int half16 = lane & 15;
  int hi  = (lane >= 16);
  int kb0 = hi ? 16 : 0, kb1 = hi ? 24 : 8;
  int kbB = hi ? 16 : 0;

  const u16* arow  = A  + (size_t)(m0 + half16) * Kd;
  const u16* brow0 = Bt + (size_t)(n0 + half16) * Kd + kbB;

  v8f acc0 = {}, acc1 = {}, acc2 = {}, acc3 = {};
  for (int kk = 0; kk < Kd; kk += 32) {
    __builtin_prefetch(arow + kk + 256, 0, 3);
    v16bf a  = load_frag_a(arow + kk, kb0, kb1);
    v16bf b0 = load_frag_b(brow0 + kk);
    v16bf b1 = load_frag_b(brow0 + (size_t)16 * Kd + kk);
    v16bf b2 = load_frag_b(brow0 + (size_t)32 * Kd + kk);
    v16bf b3 = load_frag_b(brow0 + (size_t)48 * Kd + kk);
    acc0 = wmma_bf16(a, b0, acc0);
    acc1 = wmma_bf16(a, b1, acc1);
    acc2 = wmma_bf16(a, b2, acc2);
    acc3 = wmma_bf16(a, b3, acc3);
  }
  int mrow = m0 + (hi ? 8 : 0);
#pragma unroll
  for (int r = 0; r < 8; ++r) {
    float* crow = C + (size_t)(mrow + r) * N + n0 + half16;
    crow[0]  = acc0[r];
    crow[16] = acc1[r];
    crow[32] = acc2[r];
    crow[48] = acc3[r];
  }
}

// ---------------------------------------------------------------------------
// Flash attention. One wave = (head h, 16-query tile qt). Key blocks of 32.
// Qr [h][s][128] (scaled), Kr [kv][s][128], Vt [kv*128 + d][s]  (all bf16)
// attB out: [s][2048] bf16.
// Block mapping: all 8 waves of a block share the same qt (same trip count).
// ---------------------------------------------------------------------------
__global__ void k_attn(const u16* __restrict__ Qr, const u16* __restrict__ Kr,
                       const u16* __restrict__ Vt, u16* __restrict__ attB) {
  __shared__ __align__(16) u16 pl[8][16 * 32];   // per-wave P transpose tile

  int widb = threadIdx.x >> 5;
  int lane = threadIdx.x & 31;
  int h  = ((blockIdx.x & 1) << 3) + widb;       // 0..15
  int qt = blockIdx.x >> 1;                      // 0..255
  int hkv = h >> 2;
  int qbase = qt << 4;
  int half16 = lane & 15;
  int hi  = (lane >= 16);
  int kb0 = hi ? 16 : 0, kb1 = hi ? 24 : 8;
  int kbB = hi ? 16 : 0;

  // Q A-fragments across HD=128 (4 k-steps of 32), kept resident
  const u16* qrow = Qr + ((size_t)h * S_LEN + qbase + half16) * HEAD_DIM;
  v16bf qa[4];
#pragma unroll
  for (int ks = 0; ks < 4; ++ks) qa[ks] = load_frag_a(qrow + ks * 32, kb0, kb1);

  v8f O[8];
#pragma unroll
  for (int nt = 0; nt < 8; ++nt) O[nt] = (v8f){};
  float mrun[8], lrun[8];
#pragma unroll
  for (int r = 0; r < 8; ++r) { mrun[r] = -1e30f; lrun[r] = 0.0f; }

  const u16* kbase_ptr = Kr + (size_t)hkv * S_LEN * HEAD_DIM;
  const u16* vbase_ptr = Vt + (size_t)hkv * HEAD_DIM * S_LEN;
  int nblk = ((qbase + 15) >> 5) + 1;            // causal trip count
  int qrow_m = qbase + (hi ? 8 : 0);

  for (int b = 0; b < nblk; ++b) {
    int k0 = b << 5;
    // ---- scores: 16x32 via two 16x16 WMMA accumulations over HD ----
    v8f s0 = {}, s1 = {};
    const u16* krow0 = kbase_ptr + (size_t)(k0 + half16) * HEAD_DIM + kbB;
    const u16* krow1 = krow0 + 16 * HEAD_DIM;
    __builtin_prefetch(krow0 + 32 * HEAD_DIM, 0, 3);
#pragma unroll
    for (int ks = 0; ks < 4; ++ks) {
      s0 = wmma_bf16(qa[ks], load_frag_b(krow0 + ks * 32), s0);
      s1 = wmma_bf16(qa[ks], load_frag_b(krow1 + ks * 32), s1);
    }
    // ---- causal mask + online softmax (rows live in 16-lane groups) ----
    int key0 = k0 + half16, key1 = key0 + 16;
#pragma unroll
    for (int r = 0; r < 8; ++r) {
      int qm = qrow_m + r;
      float v0 = s0[r] + (key0 > qm ? -1e9f : 0.0f);
      float v1 = s1[r] + (key1 > qm ? -1e9f : 0.0f);
      float t = fmaxf(v0, v1);
      t = fmaxf(t, __shfl_xor(t, 1, 32));
      t = fmaxf(t, __shfl_xor(t, 2, 32));
      t = fmaxf(t, __shfl_xor(t, 4, 32));
      t = fmaxf(t, __shfl_xor(t, 8, 32));
      float mnew  = fmaxf(mrun[r], t);
      float alpha = __expf(mrun[r] - mnew);
      float p0 = __expf(v0 - mnew);
      float p1 = __expf(v1 - mnew);
      float ps = p0 + p1;
      ps += __shfl_xor(ps, 1, 32);
      ps += __shfl_xor(ps, 2, 32);
      ps += __shfl_xor(ps, 4, 32);
      ps += __shfl_xor(ps, 8, 32);
      lrun[r] = lrun[r] * alpha + ps;
      mrun[r] = mnew;
#pragma unroll
      for (int nt = 0; nt < 8; ++nt) O[nt][r] *= alpha;
      int m = r + (hi ? 8 : 0);
      pl[widb][m * 32 + half16]      = f2b(p0);
      pl[widb][m * 32 + 16 + half16] = f2b(p1);
    }
    // ---- P (C-layout) -> A-fragment via LDS transpose ----
    const u16* prow = &pl[widb][half16 * 32];
    v16bf pa = load_frag_a(prow, kb0, kb1);
    // ---- O += P @ V  (8 hd tiles of 16) ----
    const u16* vcol = vbase_ptr + (size_t)half16 * S_LEN + k0 + kbB;
#pragma unroll
    for (int nt = 0; nt < 8; ++nt)
      O[nt] = wmma_bf16(pa, load_frag_b(vcol + (size_t)nt * 16 * S_LEN), O[nt]);
  }

  // ---- normalize and store bf16 [s][h*128 + d] ----
#pragma unroll
  for (int r = 0; r < 8; ++r) {
    float inv = 1.0f / lrun[r];
    u16* orow = attB + (size_t)(qrow_m + r) * NQ + h * HEAD_DIM + half16;
#pragma unroll
    for (int nt = 0; nt < 8; ++nt) orow[nt * 16] = f2b(O[nt][r] * inv);
  }
}

// ---------------------------------------------------------------------------
// Host launcher
// ---------------------------------------------------------------------------
extern "C" void kernel_launch(void* const* d_in, const int* in_sizes, int n_in,
                              void* d_out, int out_size, void* d_ws, size_t ws_size,
                              hipStream_t stream) {
  const float* X  = (const float*)d_in[0];
  const float* Wq = (const float*)d_in[1];
  const float* Wk = (const float*)d_in[2];
  const float* Wv = (const float*)d_in[3];
  const float* Wo = (const float*)d_in[4];
  // d_in[5]: attention_mask — exactly causal(-1e9); applied analytically.
  const int* pos  = (const int*)d_in[6];
  float* out = (float*)d_out;

  char* w = (char*)d_ws;
  u16* Xb   = (u16*)w;  w += (size_t)S_LEN * D_MODEL * 2;               // 16 MB
  u16* Wqt  = (u16*)w;  w += (size_t)NQ * D_MODEL * 2;                  //  8 MB
  u16* Wkt  = (u16*)w;  w += (size_t)NKV * D_MODEL * 2;                 //  2 MB
  u16* Wvt  = (u16*)w;  w += (size_t)NKV * D_MODEL * 2;                 //  2 MB
  u16* Wot  = (u16*)w;  w += (size_t)D_MODEL * NQ * 2;                  //  8 MB
  u16* Qr   = (u16*)w;  w += (size_t)N_HEADS * S_LEN * HEAD_DIM * 2;    // 16 MB
  u16* Kr   = (u16*)w;  w += (size_t)N_KV * S_LEN * HEAD_DIM * 2;       //  4 MB
  u16* Vt   = (u16*)w;  w += (size_t)N_KV * HEAD_DIM * S_LEN * 2;       //  4 MB
  u16* attB = (u16*)w;  w += (size_t)S_LEN * NQ * 2;                    // 16 MB

  const int T = 256;
  // 1) convert inputs to bf16 (weights transposed for contiguous B columns)
  {
    int n = S_LEN * D_MODEL;
    k_f32_to_bf16<<<(n + T - 1) / T, T, 0, stream>>>(X, Xb, n);
  }
  k_transpose_bf16<<<(D_MODEL * NQ  + T - 1) / T, T, 0, stream>>>(Wq, Wqt, D_MODEL, NQ);
  k_transpose_bf16<<<(D_MODEL * NKV + T - 1) / T, T, 0, stream>>>(Wk, Wkt, D_MODEL, NKV);
  k_transpose_bf16<<<(D_MODEL * NKV + T - 1) / T, T, 0, stream>>>(Wv, Wvt, D_MODEL, NKV);
  k_transpose_bf16<<<(NQ * D_MODEL  + T - 1) / T, T, 0, stream>>>(Wo, Wot, NQ, D_MODEL);

  // 2) fused QKV projections (WMMA 16x128 wave tile + RoPE / V-transpose)
  {
    int waves = (S_LEN / 16) * N_HEADS;     // 4096
    k_gemm_qkv<<<waves / 8, 256, 0, stream>>>(Xb, Wqt, pos, Qr, S_LEN, N_HEADS, D_MODEL, 0);
  }
  {
    int waves = (S_LEN / 16) * N_KV;        // 1024
    k_gemm_qkv<<<waves / 8, 256, 0, stream>>>(Xb, Wkt, pos, Kr, S_LEN, N_KV, D_MODEL, 1);
    k_gemm_qkv<<<waves / 8, 256, 0, stream>>>(Xb, Wvt, pos, Vt, S_LEN, N_KV, D_MODEL, 2);
  }

  // 3) flash attention (WMMA): 16 heads * 256 q-tiles = 4096 waves
  k_attn<<<512, 256, 0, stream>>>(Qr, Kr, Vt, attB);

  // 4) output projection -> f32 d_out (WMMA)
  {
    int waves = (S_LEN / 16) * (D_MODEL / 64);
    k_gemm_bf16<<<waves / 8, 256, 0, stream>>>(attB, Wot, out, S_LEN, D_MODEL, NQ);
  }
}
